// CRF_41034117546012
// MI455X (gfx1250) — compile-verified, matrix operand order checked by reading, not run on hardware
//
#include <hip/hip_runtime.h>

// CRF forward (log-space scan) on gfx1250 via V_WMMA_F32_16X16X32_F16.
//
//   alpha[t,b,j] = x[b,t,j] + m_b + log( sum_k expT[j,k] * exp(alpha[t-1,b,k]-m_b) )
//
// Mapping: D[j,b] = A(expT rows, constant) x B(p^T).  CDNA5 B- and D-fragment
// layouts share lane orientation (N across lanes), so the D->B recurrence is
// per-lane only: lane b holds j/k in {0..7,16..23}, lane b+16 holds {8..15,24..31}.
// All transcendentals use the raw base-2 HW ops (v_exp_f32 / v_log_f32) with the
// base conversion folded into FMAs; S >= 1 always, so no denorm guards needed.

typedef __attribute__((ext_vector_type(16))) _Float16 v16h;
typedef __attribute__((ext_vector_type(8)))  float    v8f;

namespace {
constexpr int kB = 1024;
constexpr int kT = 512;
constexpr int kC = 32;
constexpr int kTilesPerWave = 2;                  // two independent chains per wave (ILP)
constexpr int kWaves = (kB / 16) / kTilesPerWave; // 32 waves
constexpr float kLog2e = 1.44269504088896340736f; // log2(e)
constexpr float kLn2   = 0.69314718055994530942f; // ln(2)
}

__global__ __launch_bounds__(128, 1)
void crf_forward_wmma(const float* __restrict__ pad_x,  // [B, T, C]
                      const float* __restrict__ trans,  // [C, C] = [j, k]
                      const float* __restrict__ orig,   // [C]
                      float* __restrict__ out)          // [T, B, C]
{
  const int lane = threadIdx.x & 31;
  const int wave = (int)(blockIdx.x * (blockDim.x >> 5) + (threadIdx.x >> 5));
  if (wave >= kWaves) return;   // uniform per wave: never diverges within a wave

  const int mrow = lane & 15;          // A-role: j%16 ; B/D-role: batch-in-tile
  const int sel  = (lane >> 4) << 3;   // 0 for lanes 0-15, 8 for lanes 16-31
  const int j0   = sel;                // this lane owns j/k = j0..j0+7 and j1..j1+7
  const int j1   = 16 + sel;

  // ---- Constant A fragments: expT rows. A0 = j 0..15, A1 = j 16..31.
  // 16-bit A 16x32 layout: lanes hold M=j%16; half h -> K = (h<8 ? h : 8+h) + sel.
  v16h a0, a1;
#pragma unroll
  for (int h = 0; h < 16; ++h) {
    const int k = ((h < 8) ? h : (8 + h)) + sel;
    a0[h] = (_Float16)__builtin_amdgcn_exp2f(trans[mrow * kC + k] * kLog2e);
    a1[h] = (_Float16)__builtin_amdgcn_exp2f(trans[(16 + mrow) * kC + k] * kLog2e);
  }

  int   bidx[kTilesPerWave];
  float acur[kTilesPerWave][16];       // alpha for my lane's (batch, 16 tags) slots

  // ---- t = 0 : alpha0 = x[:,0,:] + orig  (also store)
#pragma unroll
  for (int c = 0; c < kTilesPerWave; ++c) {
    const int b = (wave * kTilesPerWave + c) * 16 + mrow;
    bidx[c] = b;
    const float4* xr = (const float4*)(pad_x + ((size_t)b * kT + 0) * kC);
    const float4* og = (const float4*)orig;
    const float4 xv0 = xr[(j0 >> 2)], xv1 = xr[(j0 >> 2) + 1];
    const float4 xv2 = xr[(j1 >> 2)], xv3 = xr[(j1 >> 2) + 1];
    const float4 ov0 = og[(j0 >> 2)], ov1 = og[(j0 >> 2) + 1];
    const float4 ov2 = og[(j1 >> 2)], ov3 = og[(j1 >> 2) + 1];
    acur[c][0]  = xv0.x + ov0.x;  acur[c][1]  = xv0.y + ov0.y;
    acur[c][2]  = xv0.z + ov0.z;  acur[c][3]  = xv0.w + ov0.w;
    acur[c][4]  = xv1.x + ov1.x;  acur[c][5]  = xv1.y + ov1.y;
    acur[c][6]  = xv1.z + ov1.z;  acur[c][7]  = xv1.w + ov1.w;
    acur[c][8]  = xv2.x + ov2.x;  acur[c][9]  = xv2.y + ov2.y;
    acur[c][10] = xv2.z + ov2.z;  acur[c][11] = xv2.w + ov2.w;
    acur[c][12] = xv3.x + ov3.x;  acur[c][13] = xv3.y + ov3.y;
    acur[c][14] = xv3.z + ov3.z;  acur[c][15] = xv3.w + ov3.w;

    float4* op = (float4*)(out + ((size_t)0 * kB + b) * kC);
    op[(j0 >> 2)]     = make_float4(acur[c][0],  acur[c][1],  acur[c][2],  acur[c][3]);
    op[(j0 >> 2) + 1] = make_float4(acur[c][4],  acur[c][5],  acur[c][6],  acur[c][7]);
    op[(j1 >> 2)]     = make_float4(acur[c][8],  acur[c][9],  acur[c][10], acur[c][11]);
    op[(j1 >> 2) + 1] = make_float4(acur[c][12], acur[c][13], acur[c][14], acur[c][15]);
  }

  // ---- sequential scan over t, two independent chains interleaved
  for (int t = 1; t < kT; ++t) {
    float xv[kTilesPerWave][16];
#pragma unroll
    for (int c = 0; c < kTilesPerWave; ++c) {
      const float4* xr = (const float4*)(pad_x + ((size_t)bidx[c] * kT + t) * kC);
      const float4 r0 = xr[(j0 >> 2)], r1 = xr[(j0 >> 2) + 1];
      const float4 r2 = xr[(j1 >> 2)], r3 = xr[(j1 >> 2) + 1];
      xv[c][0]  = r0.x; xv[c][1]  = r0.y; xv[c][2]  = r0.z; xv[c][3]  = r0.w;
      xv[c][4]  = r1.x; xv[c][5]  = r1.y; xv[c][6]  = r1.z; xv[c][7]  = r1.w;
      xv[c][8]  = r2.x; xv[c][9]  = r2.y; xv[c][10] = r2.z; xv[c][11] = r2.w;
      xv[c][12] = r3.x; xv[c][13] = r3.y; xv[c][14] = r3.z; xv[c][15] = r3.w;
    }

#pragma unroll
    for (int c = 0; c < kTilesPerWave; ++c) {
      // row max over all 32 tags of batch b (other 16 live in lane^16)
      float m = acur[c][0];
#pragma unroll
      for (int i = 1; i < 16; ++i) m = fmaxf(m, acur[c][i]);
      m = fmaxf(m, __shfl_xor(m, 16, 32));

      // B fragment: half h holds p for k = (h<8 ? h : 8+h) + sel -- identity with
      // acur's slot ordering, so purely per-lane:  p = 2^(alpha*log2e - m*log2e).
      const float nm = -m * kLog2e;
      v16h bb;
#pragma unroll
      for (int h = 0; h < 16; ++h)
        bb[h] = (_Float16)__builtin_amdgcn_exp2f(fmaf(acur[c][h], kLog2e, nm));

      v8f c0 = {}, c1 = {};
      c0 = __builtin_amdgcn_wmma_f32_16x16x32_f16(false, a0, false, bb,
                                                  (short)0, c0, false, false);
      c1 = __builtin_amdgcn_wmma_f32_16x16x32_f16(false, a1, false, bb,
                                                  (short)0, c1, false, false);

      // alpha_new = (x + m) + ln2*log2(S); S in [1, 32e] so raw v_log_f32 is safe.
      // D0 reg i -> j = i + sel, D1 reg i -> j = 16 + i + sel.
#pragma unroll
      for (int i = 0; i < 8; ++i) {
        acur[c][i]     = fmaf(kLn2, __builtin_amdgcn_logf(c0[i]), xv[c][i]     + m);
        acur[c][8 + i] = fmaf(kLn2, __builtin_amdgcn_logf(c1[i]), xv[c][8 + i] + m);
      }

      float4* op = (float4*)(out + ((size_t)t * kB + bidx[c]) * kC);
      op[(j0 >> 2)]     = make_float4(acur[c][0],  acur[c][1],  acur[c][2],  acur[c][3]);
      op[(j0 >> 2) + 1] = make_float4(acur[c][4],  acur[c][5],  acur[c][6],  acur[c][7]);
      op[(j1 >> 2)]     = make_float4(acur[c][8],  acur[c][9],  acur[c][10], acur[c][11]);
      op[(j1 >> 2) + 1] = make_float4(acur[c][12], acur[c][13], acur[c][14], acur[c][15]);
    }
  }
}

extern "C" void kernel_launch(void* const* d_in, const int* in_sizes, int n_in,
                              void* d_out, int out_size, void* d_ws, size_t ws_size,
                              hipStream_t stream) {
  (void)in_sizes; (void)n_in; (void)out_size; (void)d_ws; (void)ws_size;
  const float* pad_x = (const float*)d_in[0];   // [B, T, C] f32
  const float* trans = (const float*)d_in[1];   // [C, C]    f32
  const float* orig  = (const float*)d_in[2];   // [C]       f32
  // d_in[3] = batch_sizes (unused by the math)
  float* out = (float*)d_out;                   // [T, B, C] f32

  const int threads = 128;                                    // 4 waves/block
  const int blocks  = (kWaves * 32 + threads - 1) / threads;  // 8 blocks
  hipLaunchKernelGGL(crf_forward_wmma, dim3(blocks), dim3(threads), 0, stream,
                     pad_x, trans, orig, out);
}
